// QueryGuidedRouter_40312563040753
// MI455X (gfx1250) — matrix operand order repaired and checked by reading, not compiled, not run on hardware
//
#include <hip/hip_runtime.h>
#include <hip/hip_fp16.h>
#include <math.h>

// QueryGuidedRouter fused kernel for MI455X (gfx1250, wave32, WMMA).
// B=65536 tokens, D=H=512, E=64, TOPK=2.
// - Weights are pre-converted fp32->f16 into d_ws once per launch (L2-resident,
//   2.3MB) so the GEMM inner loop is pure ds_load/global_load_b128 + v_wmma.
// - 32 tokens per 256-thread block; each wave computes one 16-wide N tile for
//   both 16-row M tiles (one B fragment feeds two WMMAs).
// - All five GEMMs chain through LDS f16 activations; epilogue does
//   softmax/top-2/renormalize and coalesced output writes.

typedef __attribute__((ext_vector_type(16))) _Float16 v16h;
typedef __attribute__((ext_vector_type(8)))  _Float16 v8h;
typedef __attribute__((ext_vector_type(8)))  float    v8f;
typedef __attribute__((ext_vector_type(4)))  float    v4f;

#define D_IN   512
#define H_DIM  512
#define E_DIM  64
#define G4     256   // 4*E
#define TM     32    // tokens per block (2 WMMA M tiles)
#define NW     8     // waves per block (256 threads / wave32)

// f16 weight workspace layout (element offsets)
#define OFF_QE1 0u
#define OFF_QE2 262144u          // 512*512
#define OFF_FUS 524288u          // + 512*512
#define OFF_G1  1048576u         // + 512*1024
#define OFF_G2  1179648u         // + 256*512
#define W16_ELEMS 1196032u       // + 64*256
#define WS_NEEDED (W16_ELEMS * 2u)

// --- A fragment: 16x32 f16 tile from LDS, contiguous-K order (B matches) -----------------
// lanes 0-15 : M=lane,    K = kbase + 0..15
// lanes 16-31: M=lane-16, K = kbase + 16..31
__device__ __forceinline__ v16h load_a_frag(const _Float16* s, int stride,
                                            int kbase, int lane) {
  const int m  = lane & 15;
  const int hk = (lane >> 4) * 16;
  const _Float16* p = s + m * stride + kbase + hk;
  v8h lo = *(const v8h*)(p);
  v8h hi = *(const v8h*)(p + 8);
  v16h a;
#pragma unroll
  for (int i = 0; i < 8; ++i) { a[i] = lo[i]; a[i + 8] = hi[i]; }
  return a;
}

// --- B fragment: 32x16 (KxN) from row-major W[N][K], contiguous-K order (A matches) ------
// lanes 0-15 : N=nbase+lane,    K = kbase + 0..15
// lanes 16-31: N=nbase+lane-16, K = kbase + 16..31
__device__ __forceinline__ v16h load_b_frag(const float* __restrict__ W, int K,
                                            int nbase, int kbase, int lane) {
  const int n  = nbase + (lane & 15);
  const int kk = kbase + (lane >> 4) * 16;
  const float* p = W + (size_t)n * K + kk;
  v4f f0 = *(const v4f*)(p);
  v4f f1 = *(const v4f*)(p + 4);
  v4f f2 = *(const v4f*)(p + 8);
  v4f f3 = *(const v4f*)(p + 12);
  v16h b;
#pragma unroll
  for (int i = 0; i < 4; ++i) {
    b[i]      = (_Float16)f0[i];
    b[i + 4]  = (_Float16)f1[i];
    b[i + 8]  = (_Float16)f2[i];
    b[i + 12] = (_Float16)f3[i];
  }
  return b;
}

__device__ __forceinline__ v16h load_b_frag(const _Float16* __restrict__ W, int K,
                                            int nbase, int kbase, int lane) {
  const int n  = nbase + (lane & 15);
  const int kk = kbase + (lane >> 4) * 16;
  const _Float16* p = W + (size_t)n * K + kk;
  v8h lo = *(const v8h*)(p);
  v8h hi = *(const v8h*)(p + 8);
  v16h b;
#pragma unroll
  for (int i = 0; i < 8; ++i) { b[i] = lo[i]; b[i + 8] = hi[i]; }
  return b;
}

// Dual-M GEMM: one B fragment feeds two 16x16 WMMA tiles (rows 0-15, 16-31).
template <typename WT>
__device__ __forceinline__ void gemm_tile2(const _Float16* sA, int strideA,
                                           const WT* __restrict__ W, int K,
                                           int nbase, int lane,
                                           v8f& acc0, v8f& acc1) {
  for (int kb = 0; kb < K; kb += 32) {
    v16h b  = load_b_frag(W, K, nbase, kb, lane);
    v16h a0 = load_a_frag(sA, strideA, kb, lane);
    v16h a1 = load_a_frag(sA + 16 * strideA, strideA, kb, lane);
    acc0 = __builtin_amdgcn_wmma_f32_16x16x32_f16(false, a0, false, b,
                                                  (short)0, acc0, false, false);
    acc1 = __builtin_amdgcn_wmma_f32_16x16x32_f16(false, a1, false, b,
                                                  (short)0, acc1, false, false);
  }
}

// D layout: lanes 0-15 -> N=lane, rows M=0..7 in c[0..7];
//           lanes 16-31 -> N=lane-16, rows M=8..15.
// act: 0 = +bias, 1 = relu(+bias), 2 = tanh (no bias)
__device__ __forceinline__ void store_tile_f16(_Float16* sO, int strideO,
                                               int mofs, int nbase, int lane,
                                               v8f c,
                                               const float* __restrict__ bias,
                                               int act) {
  const int n  = nbase + (lane & 15);
  const int mb = mofs + (lane >> 4) * 8;
  const float bv = bias ? bias[n] : 0.0f;
#pragma unroll
  for (int r = 0; r < 8; ++r) {
    float v = c[r] + bv;
    if (act == 1) v = fmaxf(v, 0.0f);
    if (act == 2) v = tanhf(v);
    sO[(mb + r) * strideO + n] = (_Float16)v;
  }
}

template <typename WT>
__global__ __launch_bounds__(256)
void QueryGuidedRouter_fused(const float* __restrict__ mm,
                             const float* __restrict__ qf,
                             const WT* __restrict__ Wqe1,
                             const float* __restrict__ bqe1,
                             const WT* __restrict__ Wqe2,
                             const float* __restrict__ bqe2,
                             const WT* __restrict__ Wfus,
                             const float* __restrict__ bfus,
                             const WT* __restrict__ Wg1,
                             const WT* __restrict__ Wg2,
                             float* __restrict__ comb,
                             float* __restrict__ lg_out,
                             float* __restrict__ ew_out,
                             int*   __restrict__ idx_out) {
  __shared__ _Float16 sA[TM * 1024];  // qf  -> (mm | q2)   (GEMM1 A / GEMM3 A)
  __shared__ _Float16 sBf[TM * 512];  // h1  -> fused       (GEMM2 A / GEMM4 A)
  __shared__ _Float16 sT[TM * 256];   // tanh acts          (GEMM5 A)
  __shared__ float    sL[TM * 64];    // logits
  __shared__ float sMx[TM], sInv[TM], sW1[TM], sW2[TM];
  __shared__ int   sI1[TM], sI2[TM];

  const int tid  = threadIdx.x;
  const int wave = tid >> 5;
  const int lane = tid & 31;
  const long tok0 = (long)blockIdx.x * TM;

  // Pull the (L2-resident) weight panels toward the WGP early.
  __builtin_prefetch(Wqe1 + (size_t)tid * 32, 0, 1);
  __builtin_prefetch(Wfus + (size_t)tid * 32, 0, 1);

  // Stage query feats fp32 -> f16 into sA[:, 0:512] (stride 1024).
  for (int i = tid; i < TM * 128; i += 256) {
    const int r = i >> 7, c = (i & 127) * 4;
    v4f q = *(const v4f*)(qf + (tok0 + r) * D_IN + c);
#pragma unroll
    for (int j = 0; j < 4; ++j) sA[r * 1024 + c + j] = (_Float16)q[j];
  }
  __syncthreads();

  // GEMM1: h1 = relu(qf @ Wqe1.T + b1)            [32,512] K=512 -> sBf
  for (int nt = wave; nt < H_DIM / 16; nt += NW) {
    v8f a0 = {}, a1 = {};
    gemm_tile2(sA, 1024, Wqe1, D_IN, nt * 16, lane, a0, a1);
    store_tile_f16(sBf, 512, 0,  nt * 16, lane, a0, bqe1, 1);
    store_tile_f16(sBf, 512, 16, nt * 16, lane, a1, bqe1, 1);
  }
  __syncthreads();

  // Stage mm fp32 -> f16 into sA[:, 0:512] (qf no longer needed), and
  // GEMM2: q2 = h1 @ Wqe2.T + b2 -> sA[:, 512:1024]. Disjoint columns.
  for (int i = tid; i < TM * 128; i += 256) {
    const int r = i >> 7, c = (i & 127) * 4;
    v4f m = *(const v4f*)(mm + (tok0 + r) * H_DIM + c);
#pragma unroll
    for (int j = 0; j < 4; ++j) sA[r * 1024 + c + j] = (_Float16)m[j];
  }
  for (int nt = wave; nt < H_DIM / 16; nt += NW) {
    v8f a0 = {}, a1 = {};
    gemm_tile2(sBf, 512, Wqe2, H_DIM, nt * 16, lane, a0, a1);
    store_tile_f16(sA, 1024, 0,  512 + nt * 16, lane, a0, bqe2, 0);
    store_tile_f16(sA, 1024, 16, 512 + nt * 16, lane, a1, bqe2, 0);
  }
  __syncthreads();

  // GEMM3: fused = relu(sA @ Wfus.T + bfus)       [32,512] K=1024 -> sBf
  for (int nt = wave; nt < H_DIM / 16; nt += NW) {
    v8f a0 = {}, a1 = {};
    gemm_tile2(sA, 1024, Wfus, 1024, nt * 16, lane, a0, a1);
    store_tile_f16(sBf, 512, 0,  nt * 16, lane, a0, bfus, 1);
    store_tile_f16(sBf, 512, 16, nt * 16, lane, a1, bfus, 1);
  }
  __syncthreads();

  // GEMM4: t = tanh(fused @ Wg1.T)                [32,256] K=512 -> sT
  for (int nt = wave; nt < G4 / 16; nt += NW) {
    v8f a0 = {}, a1 = {};
    gemm_tile2(sBf, 512, Wg1, H_DIM, nt * 16, lane, a0, a1);
    store_tile_f16(sT, G4, 0,  nt * 16, lane, a0, nullptr, 2);
    store_tile_f16(sT, G4, 16, nt * 16, lane, a1, nullptr, 2);
  }
  __syncthreads();

  // GEMM5: logits = t @ Wg2.T (fp32 out)          [32,64] K=256 -> sL
  for (int nt = wave; nt < E_DIM / 16; nt += NW) {
    v8f a0 = {}, a1 = {};
    gemm_tile2(sT, G4, Wg2, G4, nt * 16, lane, a0, a1);
    const int n  = nt * 16 + (lane & 15);
    const int mb = (lane >> 4) * 8;
#pragma unroll
    for (int r = 0; r < 8; ++r) {
      sL[(mb + r) * 64 + n]      = a0[r];
      sL[(16 + mb + r) * 64 + n] = a1[r];
    }
  }
  __syncthreads();

  // Per-token softmax + top-2 + renormalize (one thread per token).
  if (tid < TM) {
    const float* l = sL + tid * 64;
    float mx = -3.4e38f;
    for (int e = 0; e < 64; ++e) mx = fmaxf(mx, l[e]);
    float s = 0.0f;
    for (int e = 0; e < 64; ++e) s += __expf(l[e] - mx);
    const float inv = 1.0f / s;
    float w1 = -1.0f, w2 = -1.0f; int i1 = 0, i2 = 0;
    for (int e = 0; e < 64; ++e) {
      const float we = __expf(l[e] - mx) * inv;
      if (we > w1)      { w2 = w1; i2 = i1; w1 = we; i1 = e; }
      else if (we > w2) { w2 = we; i2 = e; }
    }
    const float dn = 1.0f / (w1 + w2 + 1e-6f);
    sMx[tid] = mx; sInv[tid] = inv;
    sI1[tid] = i1; sI2[tid] = i2;
    sW1[tid] = w1 * dn; sW2[tid] = w2 * dn;
  }
  __syncthreads();

  // Coalesced writes of logits / expert_weights / combine_weights.
  for (int i = tid; i < TM * 64; i += 256) {
    const int m = i >> 6, e = i & 63;
    const float lg = sL[i];
    const float we = __expf(lg - sMx[m]) * sInv[m];
    const float cw = (e == sI1[m]) ? sW1[m] : (e == sI2[m]) ? sW2[m] : 0.0f;
    const long o = (tok0 + m) * 64 + e;
    lg_out[o] = lg;
    ew_out[o] = we;
    comb[o]   = cw;
  }
  if (tid < TM * 2) {
    const int m = tid >> 1, k = tid & 1;
    idx_out[(tok0 + m) * 2 + k] = k ? sI2[m] : sI1[m];
  }
}

// One-shot weight conversion fp32 -> f16 (weights are launch-invariant).
__global__ __launch_bounds__(256)
void cvt_f32_to_f16(const float* __restrict__ src, _Float16* __restrict__ dst,
                    int n) {
  const int i = (blockIdx.x * 256 + threadIdx.x) * 8;
  if (i + 8 <= n) {
    v4f a = *(const v4f*)(src + i);
    v4f b = *(const v4f*)(src + i + 4);
    v8h o;
#pragma unroll
    for (int j = 0; j < 4; ++j) {
      o[j]     = (_Float16)a[j];
      o[j + 4] = (_Float16)b[j];
    }
    *(v8h*)(dst + i) = o;
  }
}

extern "C" void kernel_launch(void* const* d_in, const int* in_sizes, int n_in,
                              void* d_out, int out_size, void* d_ws, size_t ws_size,
                              hipStream_t stream) {
  const float* mm   = (const float*)d_in[0];  // multimodal_feat [B,H]
  const float* qf   = (const float*)d_in[1];  // query_feat      [B,D]
  const float* Wqe1 = (const float*)d_in[2];  // [H,D]
  const float* bqe1 = (const float*)d_in[3];  // [H]
  const float* Wqe2 = (const float*)d_in[4];  // [H,H]
  const float* bqe2 = (const float*)d_in[5];  // [H]
  const float* Wfus = (const float*)d_in[6];  // [H,2H]
  const float* bfus = (const float*)d_in[7];  // [H]
  const float* Wg1  = (const float*)d_in[8];  // [4E,H]
  const float* Wg2  = (const float*)d_in[9];  // [E,4E]

  const size_t B = (size_t)in_sizes[0] / H_DIM;

  float* out  = (float*)d_out;
  float* comb = out;                         // [B,E]
  float* lg   = out + B * E_DIM;             // [B,E]
  float* ew   = out + 2 * B * E_DIM;         // [B,E]
  int*   idx  = (int*)(out + 3 * B * E_DIM); // [B,2]

  dim3 grid((unsigned)(B / TM));
  dim3 block(256);

  if (ws_size >= (size_t)WS_NEEDED) {
    // Convert weights fp32 -> f16 into workspace (deterministic, every call).
    _Float16* wh = (_Float16*)d_ws;
    struct { const float* s; unsigned o; int n; } segs[5] = {
      {Wqe1, OFF_QE1, 512 * 512},
      {Wqe2, OFF_QE2, 512 * 512},
      {Wfus, OFF_FUS, 512 * 1024},
      {Wg1,  OFF_G1,  256 * 512},
      {Wg2,  OFF_G2,  64 * 256},
    };
    for (int s = 0; s < 5; ++s) {
      const int nblk = (segs[s].n / 8 + 255) / 256;
      hipLaunchKernelGGL(cvt_f32_to_f16, dim3(nblk), block, 0, stream,
                         segs[s].s, wh + segs[s].o, segs[s].n);
    }
    hipLaunchKernelGGL((QueryGuidedRouter_fused<_Float16>), grid, block, 0, stream,
                       mm, qf,
                       wh + OFF_QE1, bqe1, wh + OFF_QE2, bqe2,
                       wh + OFF_FUS, bfus, wh + OFF_G1, wh + OFF_G2,
                       comb, lg, ew, idx);
  } else {
    // Fallback: convert weights in-flight inside the GEMM loops.
    hipLaunchKernelGGL((QueryGuidedRouter_fused<float>), grid, block, 0, stream,
                       mm, qf, Wqe1, bqe1, Wqe2, bqe2, Wfus, bfus, Wg1, Wg2,
                       comb, lg, ew, idx);
  }
}
